// MultiheadSelfAttention_48696339202515
// MI455X (gfx1250) — matrix operand order repaired and checked by reading, not compiled, not run on hardware
//
#include <hip/hip_runtime.h>

typedef __attribute__((ext_vector_type(8)))  _Float16 v8h;
typedef __attribute__((ext_vector_type(16))) _Float16 v16h;
typedef __attribute__((ext_vector_type(8)))  float    v8f;

#define NB   64          // batch
#define SP   256         // padded seq (>= 254)
#define SOUT 254         // output seq
#define EE   512         // embed
#define HH   8           // heads
#define DD   64          // head dim
#define E3   1536        // 3*E

static __device__ __forceinline__ v16h combine16(v8h a, v8h b) {
  return __builtin_shufflevector(a, b, 0,1,2,3,4,5,6,7,8,9,10,11,12,13,14,15);
}
// A-operand: two 8-half chunks at p and p+16 (per-half-wave k offset applied by caller)
static __device__ __forceinline__ v16h ldA(const _Float16* p) {
  return combine16(*(const v8h*)p, *(const v8h*)(p + 16));
}
// B-operand: two 8-half chunks at p and p+16 (per-half-wave k offset applied by caller)
static __device__ __forceinline__ v16h ldB(const _Float16* p) {
  return combine16(*(const v8h*)p, *(const v8h*)(p + 16));
}
static __device__ __forceinline__ v8f zero8() {
  v8f z = {0.f,0.f,0.f,0.f,0.f,0.f,0.f,0.f};
  return z;
}
#define WMMA(a,b,c) __builtin_amdgcn_wmma_f32_16x16x32_f16(false,(a),false,(b),(short)0,(c),false,false)

// ---------------- prep kernels ----------------
__global__ void mha_offsets_k(const int* __restrict__ agents, int* __restrict__ offs) {
  if (threadIdx.x == 0) {
    int acc = 0;
    for (int i = 0; i < NB; ++i) { offs[i] = acc; acc += agents[i]; }
  }
}

__global__ void mha_f32to16_k(const float* __restrict__ src, _Float16* __restrict__ dst, int n) {
  int i = blockIdx.x * blockDim.x + threadIdx.x;
  if (i < n) dst[i] = (_Float16)src[i];
}

// scatter packed rows -> padded f16 X [NB*SP, EE]
__global__ void mha_pad_k(const float* __restrict__ att_in, const int* __restrict__ agents,
                          const int* __restrict__ offs, _Float16* __restrict__ X) {
  int i = blockIdx.x * blockDim.x + threadIdx.x;   // over NB*SP*EE
  int e = i & (EE - 1);
  int row = i >> 9;
  int b = row >> 8, s = row & (SP - 1);
  float v = 0.f;
  if (s < agents[b]) v = att_in[(size_t)(offs[b] + s) * EE + e];
  X[i] = (_Float16)v;
}

// Vt[b][h][d][s] = qkv[b*SP+s][1024 + h*64 + d]
__global__ void mha_vt_k(const _Float16* __restrict__ qkv, _Float16* __restrict__ Vt) {
  int i = blockIdx.x * blockDim.x + threadIdx.x;   // over NB*HH*DD*SP
  int s = i & (SP - 1);
  int d = (i >> 8) & (DD - 1);
  int h = (i >> 14) & (HH - 1);
  int b = i >> 17;
  Vt[i] = qkv[(size_t)(b * SP + s) * E3 + 2 * EE + h * DD + d];
}

// ---------------- GEMM: QKV = X * Win^T + b_in  (f16 out), 32x64 tile per wave ----------
__global__ __launch_bounds__(256) void mha_gemm_qkv_k(const _Float16* __restrict__ X,
                                                      const _Float16* __restrict__ W,
                                                      const float* __restrict__ bias,
                                                      _Float16* __restrict__ out) {
  int w    = (blockIdx.x * 256 + threadIdx.x) >> 5;       // wave id == tile id
  int lane = threadIdx.x & 31;
  int half = lane >> 4, l16 = lane & 15;
  const int NT = E3 / 64;                                 // 24 column-chunks of 64
  int mt = w / NT, nc = w % NT;
  int m0 = mt * 32, n0 = nc * 64;
  int arow0 = m0 + l16, arow1 = m0 + 16 + l16;
  v8f acc[2][4] = {{zero8(), zero8(), zero8(), zero8()},
                   {zero8(), zero8(), zero8(), zero8()}};
  for (int k0 = 0; k0 < EE; k0 += 32) {
    v16h a0 = ldA(X + (size_t)arow0 * EE + k0 + (half ? 8 : 0));
    v16h a1 = ldA(X + (size_t)arow1 * EE + k0 + (half ? 8 : 0));
#pragma unroll
    for (int nt = 0; nt < 4; ++nt) {
      v16h bmat = ldB(W + (size_t)(n0 + nt * 16 + l16) * EE + k0 + (half ? 8 : 0));
      acc[0][nt] = WMMA(a0, bmat, acc[0][nt]);
      acc[1][nt] = WMMA(a1, bmat, acc[1][nt]);
    }
  }
#pragma unroll
  for (int mi = 0; mi < 2; ++mi)
#pragma unroll
    for (int nt = 0; nt < 4; ++nt) {
      int col = n0 + nt * 16 + l16;
      float bv = bias[col];
#pragma unroll
      for (int r = 0; r < 8; ++r) {
        int row = m0 + mi * 16 + r + half * 8;
        out[(size_t)row * E3 + col] = (_Float16)(acc[mi][nt][r] + bv);
      }
    }
}

// ---------------- attention: one wave per (b, h, 16-query tile) ----------------
__global__ __launch_bounds__(32) void mha_attn_k(const _Float16* __restrict__ qkv,
                                                 const _Float16* __restrict__ Vt,
                                                 const int* __restrict__ agents,
                                                 _Float16* __restrict__ ctx) {
  __shared__ float    sS[16 * SP];   // raw scores (row-major [16][256])
  __shared__ _Float16 sP[16 * SP];   // exp probs  (row-major [16][256])
  int wid = blockIdx.x;              // NB*HH*16
  int qt = wid & 15, h = (wid >> 4) & 7, b = wid >> 7;
  int lane = threadIdx.x, half = lane >> 4, l16 = lane & 15;
  int nAg = agents[b];
  const float scale = 0.125f;        // D^-0.5

  // Q A-operands for the two 32-wide d chunks
  int qrow = b * SP + qt * 16 + l16;
  const _Float16* qbase = qkv + (size_t)qrow * E3 + h * DD;
  v16h aQ0 = ldA(qbase + 0  + (half ? 8 : 0));
  v16h aQ1 = ldA(qbase + 32 + (half ? 8 : 0));

  float rmax[8];
#pragma unroll
  for (int r = 0; r < 8; ++r) rmax[r] = -3.0e38f;

  // pass 1: scores = Q K^T, scale + key mask, stream to LDS, track row max
  for (int kt = 0; kt < 16; ++kt) {
    int krow = b * SP + kt * 16 + l16;
    const _Float16* kbase = qkv + (size_t)krow * E3 + EE + h * DD + (half ? 8 : 0);
    v8f acc = zero8();
    acc = WMMA(aQ0, ldB(kbase), acc);
    acc = WMMA(aQ1, ldB(kbase + 32), acc);
    int key = kt * 16 + l16;
    bool valid = key < nAg;
#pragma unroll
    for (int r = 0; r < 8; ++r) {
      float v = valid ? acc[r] * scale : -1.0e9f;
      sS[(r + half * 8) * SP + key] = v;
      rmax[r] = fmaxf(rmax[r], v);
    }
  }
#pragma unroll
  for (int off = 1; off <= 8; off <<= 1)
#pragma unroll
    for (int r = 0; r < 8; ++r) rmax[r] = fmaxf(rmax[r], __shfl_xor(rmax[r], off, 32));

  // pass 2: exp + row sum, probs -> f16 LDS
  float rsum[8];
#pragma unroll
  for (int r = 0; r < 8; ++r) rsum[r] = 0.f;
  for (int kt = 0; kt < 16; ++kt) {
    int key = kt * 16 + l16;
#pragma unroll
    for (int r = 0; r < 8; ++r) {
      float p = __expf(sS[(r + half * 8) * SP + key] - rmax[r]);
      rsum[r] += p;
      sP[(r + half * 8) * SP + key] = (_Float16)p;
    }
  }
#pragma unroll
  for (int off = 1; off <= 8; off <<= 1)
#pragma unroll
    for (int r = 0; r < 8; ++r) rsum[r] += __shfl_xor(rsum[r], off, 32);

  // pass 3: ctx = P * V   (A from LDS, B from Vt contiguous)
  v8f accv[4] = {zero8(), zero8(), zero8(), zero8()};
  const _Float16* vb = Vt + (size_t)((b * HH + h) * DD) * SP;
  for (int kc = 0; kc < 8; ++kc) {
    v16h aP = ldA(sP + l16 * SP + kc * 32 + (half ? 8 : 0));
#pragma unroll
    for (int nt = 0; nt < 4; ++nt) {
      v16h bV = ldB(vb + (size_t)(nt * 16 + l16) * SP + kc * 32 + (half ? 8 : 0));
      accv[nt] = WMMA(aP, bV, accv[nt]);
    }
  }
  int crow0 = b * SP + qt * 16;
#pragma unroll
  for (int nt = 0; nt < 4; ++nt) {
    int d = nt * 16 + l16;
#pragma unroll
    for (int r = 0; r < 8; ++r) {
      int m = r + half * 8;
      ctx[(size_t)(crow0 + m) * EE + h * DD + d] = (_Float16)(accv[nt][r] / rsum[r]);
    }
  }
}

// ---------------- GEMM: out = ctx * Wout^T + b_out  (f32, masked, 32x64 tile) ----------
__global__ __launch_bounds__(256) void mha_gemm_out_k(const _Float16* __restrict__ A,
                                                      const _Float16* __restrict__ W,
                                                      const float* __restrict__ bias,
                                                      const int* __restrict__ agents,
                                                      float* __restrict__ out) {
  int w    = (blockIdx.x * 256 + threadIdx.x) >> 5;
  int lane = threadIdx.x & 31;
  int half = lane >> 4, l16 = lane & 15;
  const int NT = EE / 64;                                 // 8
  int mt = w / NT, nc = w % NT;
  int m0 = mt * 32, n0 = nc * 64;
  int arow0 = m0 + l16, arow1 = m0 + 16 + l16;
  v8f acc[2][4] = {{zero8(), zero8(), zero8(), zero8()},
                   {zero8(), zero8(), zero8(), zero8()}};
  for (int k0 = 0; k0 < EE; k0 += 32) {
    v16h a0 = ldA(A + (size_t)arow0 * EE + k0 + (half ? 8 : 0));
    v16h a1 = ldA(A + (size_t)arow1 * EE + k0 + (half ? 8 : 0));
#pragma unroll
    for (int nt = 0; nt < 4; ++nt) {
      v16h bmat = ldB(W + (size_t)(n0 + nt * 16 + l16) * EE + k0 + (half ? 8 : 0));
      acc[0][nt] = WMMA(a0, bmat, acc[0][nt]);
      acc[1][nt] = WMMA(a1, bmat, acc[1][nt]);
    }
  }
#pragma unroll
  for (int mi = 0; mi < 2; ++mi)
#pragma unroll
    for (int nt = 0; nt < 4; ++nt) {
      int col = n0 + nt * 16 + l16;
      float bv = bias[col];
#pragma unroll
      for (int r = 0; r < 8; ++r) {
        int row = m0 + mi * 16 + r + half * 8;
        int b = row >> 8, s = row & (SP - 1);
        if (s < SOUT) {
          float v = (s < agents[b]) ? (acc[mi][nt][r] + bv) : 0.f;
          out[(size_t)(b * SOUT + s) * EE + col] = v;
        }
      }
    }
}

// ---------------- host launcher ----------------
extern "C" void kernel_launch(void* const* d_in, const int* in_sizes, int n_in,
                              void* d_out, int out_size, void* d_ws, size_t ws_size,
                              hipStream_t stream) {
  (void)in_sizes; (void)n_in; (void)out_size; (void)ws_size;
  const float* att_in = (const float*)d_in[0];
  const float* w_in   = (const float*)d_in[1];
  const float* b_in   = (const float*)d_in[2];
  const float* w_out  = (const float*)d_in[3];
  const float* b_out  = (const float*)d_in[4];
  const int*   agents = (const int*)d_in[5];
  float* out = (float*)d_out;

  char* ws = (char*)d_ws;
  size_t off = 0;
  auto alloc = [&](size_t bytes) { void* p = ws + off; off = (off + bytes + 255) & ~(size_t)255; return p; };
  int*      offs = (int*)alloc(NB * sizeof(int));
  _Float16* Wih  = (_Float16*)alloc((size_t)E3 * EE * 2);
  _Float16* Woh  = (_Float16*)alloc((size_t)EE * EE * 2);
  _Float16* X    = (_Float16*)alloc((size_t)NB * SP * EE * 2);
  _Float16* QKV  = (_Float16*)alloc((size_t)NB * SP * E3 * 2);
  _Float16* VT   = (_Float16*)alloc((size_t)NB * HH * DD * SP * 2);
  _Float16* CTX  = (_Float16*)alloc((size_t)NB * SP * EE * 2);

  mha_offsets_k<<<1, 64, 0, stream>>>(agents, offs);
  mha_f32to16_k<<<(E3 * EE + 255) / 256, 256, 0, stream>>>(w_in, Wih, E3 * EE);
  mha_f32to16_k<<<(EE * EE + 255) / 256, 256, 0, stream>>>(w_out, Woh, EE * EE);
  mha_pad_k<<<(NB * SP * EE) / 256, 256, 0, stream>>>(att_in, agents, offs, X);

  // QKV GEMM: (NB*SP/32) m-tiles * (E3/64) n-chunks waves
  {
    int waves = (NB * SP / 32) * (E3 / 64);
    mha_gemm_qkv_k<<<waves * 32 / 256, 256, 0, stream>>>(X, Wih, b_in, QKV);
  }
  mha_vt_k<<<(NB * HH * DD * SP) / 256, 256, 0, stream>>>(QKV, VT);
  mha_attn_k<<<NB * HH * 16, 32, 0, stream>>>(QKV, VT, agents, CTX);
  {
    int waves = (NB * SP / 32) * (EE / 64);
    mha_gemm_out_k<<<waves * 32 / 256, 256, 0, stream>>>(CTX, Woh, b_out, agents, out);
  }
}
